// SelfAttention_2310692405533
// MI455X (gfx1250) — matrix operand order repaired and checked by reading, not compiled, not run on hardware
//
#include <hip/hip_runtime.h>
#include <hip/hip_bf16.h>
#include <cstdint>
#include <cstddef>

// ---------------- problem constants ----------------
constexpr int Bsz  = 2;
constexpr int Sq   = 2048;
constexpr int Dm   = 4096;
constexpr int HQ   = 32;
constexpr int HKV  = 8;
constexpr int DH   = 128;
constexpr int Mtot = Bsz * Sq;      // 4096 tokens

// ---------------- WMMA types / helpers ----------------
typedef __attribute__((ext_vector_type(16))) __bf16        bfrag_t;
typedef __attribute__((ext_vector_type(8)))  float         v8f;
typedef __attribute__((ext_vector_type(8)))  unsigned int  v8u;
typedef __attribute__((ext_vector_type(4)))  unsigned int  v4u;

__device__ inline unsigned short f2bf(float f) {
  unsigned int u = __builtin_bit_cast(unsigned int, f);
  u += 0x7FFFu + ((u >> 16) & 1u);          // round-to-nearest-even
  return (unsigned short)(u >> 16);
}
__device__ inline float bf2f(unsigned short h) {
  return __builtin_bit_cast(float, (unsigned int)h << 16);
}
__device__ inline bfrag_t mkfrag(uint4 a, uint4 b) {
  v8u t;
  t[0] = a.x; t[1] = a.y; t[2] = a.z; t[3] = a.w;
  t[4] = b.x; t[5] = b.y; t[6] = b.z; t[7] = b.w;
  return __builtin_bit_cast(bfrag_t, t);
}
__device__ inline v8f wmma_bf16(bfrag_t a, bfrag_t b, v8f c) {
  // (neg_a, A, neg_b, B, c_mod, C, reuse_a, reuse_b)
  return __builtin_amdgcn_wmma_f32_16x16x32_bf16(false, a, false, b,
                                                 (short)0, c, false, false);
}

// ---------------- Tensor Data Mover: 2-D tile load (D# per ISA ch.8) ----
// data_size = 8B units. Loads tile (tile0 units x tile1 rows) from a row-major
// tensor (row stride = stride0 units) at gptr into LDS at lds_addr.
__device__ inline void tdm_load_2d(unsigned lds_addr, const void* gptr,
                                   unsigned tile0, unsigned tile1,
                                   unsigned ten0, unsigned ten1,
                                   unsigned stride0)
{
  unsigned long long ga = (unsigned long long)(uintptr_t)gptr;
  v4u g0;
  g0[0] = 1u;                                             // count=1 (valid D#)
  g0[1] = lds_addr;                                       // LDS byte address
  g0[2] = (unsigned)ga;                                   // global_addr[31:0]
  g0[3] = ((unsigned)(ga >> 32) & 0x01FFFFFFu) | (2u << 30); // addr[56:32]|type=2
  v8u g1;
  g1[0] = 3u << 16;                                       // data_size = 8 bytes
  g1[1] = (ten0 & 0xFFFFu) << 16;                         // tensor_dim0[15:0]
  g1[2] = (ten0 >> 16) | ((ten1 & 0xFFFFu) << 16);        // dim0 hi | dim1 lo
  g1[3] = (ten1 >> 16) | ((tile0 & 0xFFFFu) << 16);       // dim1 hi | tile_dim0
  g1[4] = tile1 & 0xFFFFu;                                // tile_dim1 (tile_dim2=0)
  g1[5] = stride0;                                        // tensor_dim0_stride lo
  g1[6] = 0;
  g1[7] = 0;
  v4u gz = (v4u){0u, 0u, 0u, 0u};                         // groups 2/3 unused (2-D)
  asm volatile("tensor_load_to_lds %0, %1, %2, %3"
               :: "s"(g0), "s"(g1), "s"(gz), "s"(gz)
               : "memory");
}

// ---------------- generic WMMA GEMM ----------------
// C[M,N] = A[M,K] * B[K,N]; A f32 (ABF16=0, converted on the fly) or bf16.
// MODE 0: C f32 row-major [M,N]
// MODE 1: C bf16  [b, Hout, s, d]
// MODE 2: C bf16  [b, Hout, d, s]   (transposed V layout)
constexpr int GBM = 128, GBN = 64, GBK = 32;
constexpr int APAD = 40, BPAD = 40;   // 80B rows keep 16B alignment

template <int ABF16, int MODE>
__global__ __launch_bounds__(256) void gemm_wmma(
    const void* __restrict__ Aptr,
    const float* __restrict__ Bw,
    void* __restrict__ Cptr, int Hout,
    int Ndim, int Kdim)
{
  __shared__ __align__(16) unsigned short ldsA[GBM][APAD];
  __shared__ __align__(16) unsigned short ldsB[GBN][BPAD];

  const int tid  = threadIdx.x;
  const int lane = tid & 31;
  const int wave = tid >> 5;
  const int m0 = blockIdx.y * GBM;
  const int n0 = blockIdx.x * GBN;
  const int wm = (wave >> 1) * 32;     // 4 waves along M
  const int wn = (wave & 1) * 32;      // 2 waves along N

  v8f acc[2][2] = {};

  const int arow = tid >> 1;           // 0..127
  const int acol = (tid & 1) * 16;     // 0 or 16
  const int bk   = tid >> 3;           // 0..31
  const int bn   = (tid & 7) * 8;      // 0..56

  for (int k0 = 0; k0 < Kdim; k0 += GBK) {
    __syncthreads();
    // ---- stage A tile (128 x 32) as bf16 in LDS ----
    if (ABF16) {
      const unsigned short* Ab = (const unsigned short*)Aptr;
      const uint4* src = (const uint4*)(Ab + (size_t)(m0 + arow) * Kdim + k0 + acol);
      *(uint4*)&ldsA[arow][acol + 0] = src[0];
      *(uint4*)&ldsA[arow][acol + 8] = src[1];
    } else {
      const float* Af  = (const float*)Aptr;
      const float* src = Af + (size_t)(m0 + arow) * Kdim + k0 + acol;
      unsigned int w[8];
#pragma unroll
      for (int j = 0; j < 8; ++j)
        w[j] = (unsigned int)f2bf(src[2*j]) | ((unsigned int)f2bf(src[2*j+1]) << 16);
      *(uint4*)&ldsA[arow][acol + 0] = make_uint4(w[0], w[1], w[2], w[3]);
      *(uint4*)&ldsA[arow][acol + 8] = make_uint4(w[4], w[5], w[6], w[7]);
    }
    // ---- stage B tile (32 x 64) transposed: ldsB[n][k] ----
    {
      const float* bs = Bw + (size_t)(k0 + bk) * Ndim + n0 + bn;
#pragma unroll
      for (int j = 0; j < 8; ++j)
        ldsB[bn + j][bk] = f2bf(bs[j]);
    }
    __syncthreads();

    // ---- compute: 2x2 WMMA tiles per wave ----
    const int ah = (lane < 16) ? 0 : 8;     // A frag K-run offset
    const int bh = (lane < 16) ? 0 : 8;     // B frag K-run offset
#pragma unroll
    for (int mt = 0; mt < 2; ++mt) {
      uint4 a0 = *(const uint4*)&ldsA[wm + mt*16 + (lane & 15)][ah];
      uint4 a1 = *(const uint4*)&ldsA[wm + mt*16 + (lane & 15)][16 + ah];
      bfrag_t af = mkfrag(a0, a1);
#pragma unroll
      for (int nt = 0; nt < 2; ++nt) {
        uint4 b0 = *(const uint4*)&ldsB[wn + nt*16 + (lane & 15)][bh];
        uint4 b1 = *(const uint4*)&ldsB[wn + nt*16 + (lane & 15)][bh + 16];
        acc[mt][nt] = wmma_bf16(af, mkfrag(b0, b1), acc[mt][nt]);
      }
    }
  }

  // ---- epilogue: C/D layout row = r + 8*(lane>=16), col = lane&15 ----
  const int rowoff = (lane >> 4) * 8;
#pragma unroll
  for (int mt = 0; mt < 2; ++mt)
#pragma unroll
    for (int nt = 0; nt < 2; ++nt)
#pragma unroll
      for (int r = 0; r < 8; ++r) {
        int row = m0 + wm + mt*16 + rowoff + r;
        int col = n0 + wn + nt*16 + (lane & 15);
        float v = acc[mt][nt][r];
        if (MODE == 0) {
          ((float*)Cptr)[(size_t)row * Ndim + col] = v;
        } else {
          int b = row >> 11;       // / Sq (2048)
          int s = row & 2047;
          int h = col >> 7;        // / DH (128)
          int d = col & 127;
          unsigned short* Cb = (unsigned short*)Cptr;
          if (MODE == 1)
            Cb[(((size_t)b * Hout + h) * Sq + s) * DH + d] = f2bf(v);
          else
            Cb[(((size_t)b * Hout + h) * DH + d) * Sq + s] = f2bf(v);
        }
      }
}

// ---------------- RoPE (in place on bf16 [b,h,s,d]) ----------------
__global__ __launch_bounds__(256) void rope_kernel(
    unsigned short* __restrict__ qk, const float* __restrict__ freqs, int BH)
{
  int idx = blockIdx.x * 256 + threadIdx.x;       // over BH * Sq * 64
  int i  = idx & 63;
  int s  = (idx >> 6) & (Sq - 1);
  int bh = idx >> 17;
  if (bh >= BH) return;
  float f = freqs[s * 64 + i];
  float cs = __cosf(f), sn = __sinf(f);
  size_t base = (((size_t)bh * Sq) + s) * DH + 2 * i;
  float xe = bf2f(qk[base]);
  float xo = bf2f(qk[base + 1]);
  qk[base]     = f2bf(xe * cs - xo * sn);
  qk[base + 1] = f2bf(xe * sn + xo * cs);
}

// ---------------- flash attention with TDM-staged K/V tiles ----------------
// grid: (B*HQ, Sq/64); block: 128 (4 waves); each wave owns 16 query rows.
// All 4 waves share one (b, h_kv): K/V^T tiles are DMA'd into LDS once per
// 32-key step by wave 0 (tensor_load_to_lds, double buffered, TENSORcnt).
__global__ __launch_bounds__(128) void attn_kernel(
    const unsigned short* __restrict__ q,
    const unsigned short* __restrict__ k,
    const unsigned short* __restrict__ vt,
    unsigned short* __restrict__ attn)
{
  __shared__ __align__(16) unsigned short ldsP[4][16][32];      //  4 KB
  __shared__ __align__(16) unsigned short ldsK[2][32][DH];      // 16 KB
  __shared__ __align__(16) unsigned short ldsV[2][DH][32];      // 16 KB

  const int lane = threadIdx.x & 31;
  const int wave = threadIdx.x >> 5;
  const int bh = blockIdx.x;
  const int b  = bh >> 5;          // / HQ
  const int h  = bh & 31;
  const int hk = h >> 2;           // / N_REP
  const int q0 = blockIdx.y * 64 + wave * 16;

  const unsigned short* qbase = q  + (((size_t)(b * HQ  + h )) * Sq) * DH;
  const unsigned short* kbase = k  + (((size_t)(b * HKV + hk)) * Sq) * DH;
  const unsigned short* vbase = vt + (((size_t)(b * HKV + hk)) * DH) * Sq;

  const int ah   = (lane < 16) ? 0 : 8;    // A frag K-run offset
  const int bh16 = (lane < 16) ? 0 : 8;    // B frag K-run offset

  // preload Q fragments (16 rows x 128 d = 4 fragments of 16x32)
  bfrag_t qf[4];
  const int qrow = q0 + (lane & 15);
#pragma unroll
  for (int dc = 0; dc < 4; ++dc) {
    const unsigned short* p = qbase + (size_t)qrow * DH + dc * 32 + ah;
    qf[dc] = mkfrag(*(const uint4*)p, *(const uint4*)(p + 16));
  }

  v8f o[8] = {};
  float mrow[8], lrow[8];
#pragma unroll
  for (int r = 0; r < 8; ++r) { mrow[r] = -1e30f; lrow[r] = 0.f; }

  const float scale = 0.08838834764831845f;   // 1/sqrt(128)
  const int nkv = Sq / 32;

  // TDM prologue: tile 0 into buffer 0
  if (wave == 0) {
    // K tile: 32 key-rows of 256B (32 x 8B units), row stride DH*2 = 256B
    tdm_load_2d((unsigned)(uintptr_t)&ldsK[0][0][0], kbase,
                /*tile0*/32, /*tile1*/32, /*ten0*/32, /*ten1*/Sq, /*stride0*/32);
    // V^T tile: 128 d-rows of 64B (8 x 8B units), row stride Sq*2 bytes
    tdm_load_2d((unsigned)(uintptr_t)&ldsV[0][0][0], vbase,
                /*tile0*/8, /*tile1*/DH, /*ten0*/Sq/4, /*ten1*/DH, /*stride0*/Sq/4);
  }

  for (int it = 0; it < nkv; ++it) {
    const int buf = it & 1;
    if (wave == 0) {
      if (it + 1 < nkv) {
        const int kn = (it + 1) * 32;
        tdm_load_2d((unsigned)(uintptr_t)&ldsK[buf ^ 1][0][0],
                    kbase + (size_t)kn * DH, 32, 32, 32, Sq, 32);
        tdm_load_2d((unsigned)(uintptr_t)&ldsV[buf ^ 1][0][0],
                    vbase + kn, 8, DH, Sq / 4, DH, Sq / 4);
        __builtin_amdgcn_s_wait_tensorcnt((short)2);   // current tile done
      } else {
        __builtin_amdgcn_s_wait_tensorcnt((short)0);
      }
    }
    __syncthreads();   // publish tile `buf` to all waves

    // ---- scores: two 16x16 tiles over d=128 (4 WMMA each), K from LDS ----
    v8f s0 = {}, s1 = {};
#pragma unroll
    for (int dc = 0; dc < 4; ++dc) {
      const unsigned short* p0 = &ldsK[buf][lane & 15][dc * 32 + bh16];
      const unsigned short* p1 = &ldsK[buf][16 + (lane & 15)][dc * 32 + bh16];
      s0 = wmma_bf16(qf[dc], mkfrag(*(const uint4*)p0, *(const uint4*)(p0 + 16)), s0);
      s1 = wmma_bf16(qf[dc], mkfrag(*(const uint4*)p1, *(const uint4*)(p1 + 16)), s1);
    }

    // ---- online softmax (row reductions across 16 lanes of each half) ----
    float al[8];
#pragma unroll
    for (int r = 0; r < 8; ++r) {
      float a  = s0[r] * scale;
      float cc = s1[r] * scale;
      float mx = fmaxf(a, cc);
#pragma unroll
      for (int msk = 8; msk >= 1; msk >>= 1)
        mx = fmaxf(mx, __shfl_xor(mx, msk, 16));
      float mn    = fmaxf(mrow[r], mx);
      float alpha = __expf(mrow[r] - mn);
      mrow[r] = mn;
      a  = __expf(a  - mn);
      cc = __expf(cc - mn);
      s0[r] = a; s1[r] = cc;
      float rs = a + cc;
#pragma unroll
      for (int msk = 8; msk >= 1; msk >>= 1)
        rs += __shfl_xor(rs, msk, 16);
      lrow[r] = lrow[r] * alpha + rs;
      al[r] = alpha;
    }
#pragma unroll
    for (int c = 0; c < 8; ++c)
#pragma unroll
      for (int r = 0; r < 8; ++r)
        o[c][r] *= al[r];

    // ---- restripe P (16x32) into A-fragment layout via LDS ----
#pragma unroll
    for (int r = 0; r < 8; ++r) {
      int row = r + ((lane >> 4) << 3);
      ldsP[wave][row][lane & 15]        = f2bf(s0[r]);
      ldsP[wave][row][16 + (lane & 15)] = f2bf(s1[r]);
    }
    __syncthreads();
    const unsigned short* pp = &ldsP[wave][lane & 15][0];
    bfrag_t pf = mkfrag(*(const uint4*)(pp + ah), *(const uint4*)(pp + 16 + ah));

    // ---- O += P @ V  (8 d-chunks, V^T tile in LDS) ----
#pragma unroll
    for (int c = 0; c < 8; ++c) {
      const unsigned short* vp = &ldsV[buf][c * 16 + (lane & 15)][bh16];
      o[c] = wmma_bf16(pf, mkfrag(*(const uint4*)vp, *(const uint4*)(vp + 16)), o[c]);
    }
    __syncthreads();   // ldsP reuse + tile `buf` free for next DMA overwrite
  }

  // ---- epilogue: normalize and write attn row-major [Mtot, HQ*DH] ----
#pragma unroll
  for (int r = 0; r < 8; ++r) {
    int srow = q0 + r + ((lane >> 4) << 3);
    float inv = 1.0f / lrow[r];
#pragma unroll
    for (int c = 0; c < 8; ++c) {
      int col = h * DH + c * 16 + (lane & 15);
      attn[((size_t)(b * Sq + srow)) * (HQ * DH) + col] = f2bf(o[c][r] * inv);
    }
  }
}

// ---------------- host ----------------
extern "C" void kernel_launch(void* const* d_in, const int* in_sizes, int n_in,
                              void* d_out, int out_size, void* d_ws, size_t ws_size,
                              hipStream_t stream)
{
  (void)in_sizes; (void)n_in; (void)out_size; (void)ws_size;
  const float* x     = (const float*)d_in[0];
  const float* freqs = (const float*)d_in[1];
  const float* wq    = (const float*)d_in[2];
  const float* wk    = (const float*)d_in[3];
  const float* wv    = (const float*)d_in[4];
  const float* wo    = (const float*)d_in[5];
  // d_in[6..8] (cache_k, cache_v, start_pos=0) unused: full prefill.

  char* ws = (char*)d_ws;
  unsigned short* qb   = (unsigned short*)(ws);                              // 32 MB
  unsigned short* kb   = (unsigned short*)(ws + (size_t)32 * 1024 * 1024);   //  8 MB
  unsigned short* vtb  = (unsigned short*)(ws + (size_t)40 * 1024 * 1024);   //  8 MB
  unsigned short* attn = (unsigned short*)(ws + (size_t)48 * 1024 * 1024);   // 32 MB

  dim3 blk(256);
  // QKV projections (bf16 WMMA, f32 accumulate), layout transforms in epilogue
  gemm_wmma<0,1><<<dim3(Dm / GBN,       Mtot / GBM), blk, 0, stream>>>(x, wq, qb,  HQ,  Dm,     Dm);
  gemm_wmma<0,1><<<dim3((HKV*DH)/GBN,   Mtot / GBM), blk, 0, stream>>>(x, wk, kb,  HKV, HKV*DH, Dm);
  gemm_wmma<0,2><<<dim3((HKV*DH)/GBN,   Mtot / GBM), blk, 0, stream>>>(x, wv, vtb, HKV, HKV*DH, Dm);

  rope_kernel<<<(Bsz*HQ *Sq*64) / 256, blk, 0, stream>>>(qb, freqs, Bsz*HQ);
  rope_kernel<<<(Bsz*HKV*Sq*64) / 256, blk, 0, stream>>>(kb, freqs, Bsz*HKV);

  attn_kernel<<<dim3(Bsz*HQ, Sq/64), dim3(128), 0, stream>>>(qb, kb, vtb, attn);

  // output projection -> f32 d_out
  gemm_wmma<1,0><<<dim3(Dm / GBN, Mtot / GBM), blk, 0, stream>>>(attn, wo, (float*)d_out, 0, Dm, Dm);
}
